// Net_67095979098266
// MI455X (gfx1250) — matrix operand order repaired and checked by reading, not compiled, not run on hardware
//
#include <hip/hip_runtime.h>
#include <hip/hip_bf16.h>

typedef __attribute__((ext_vector_type(16))) __bf16 v16bf;
typedef __attribute__((ext_vector_type(8)))  float  v8f;

#define BSZ   256
#define SEQ   512
#define DIN   80
#define DPAD  96
#define HID   128
#define GATES 512     // 4*H

// ---------------- workspace byte offsets ----------------
#define WS_WEFF 0
#define WS_WHH  (GATES*DPAD*2)                 // 98304
#define WS_BEFF (WS_WHH + GATES*HID*2)         // 229376
#define WS_GX   262144                         // aligned start of Gx
#define GX_BYTES ((size_t)BSZ * SEQ * GATES * 2)   // 134217728 (bf16)

// ---------------- LDS byte offsets (recurrent kernel) ----------------
// common region first; fused-only region appended at the end
#define L_WHH   0                               // [512][128] bf16  131072
#define L_GATES 131072                          // [16][512] f32     32768
#define L_HBUF  163840                          // [16][128] bf16     4096
#define L_BIAS  167936                          // [512] f32          2048
#define L_WBR   169984                          // [4][128] f32       2048
#define L_MISC  172032                          // [8] f32              32
#define L_WEFF  172064                          // fused only: [512][96] bf16 98304
#define L_XBUF  270368                          // fused only: [2][16][96] bf16 6144
#define SMEM_GX   172064
#define SMEM_FULL 276512

// gx_gemm LDS
#define GL_WEFF 0                               // [512][96] bf16 98304
#define GL_XT   98304                           // [16][96] bf16   3072
#define SMEM_GEMM 101376

static __device__ __forceinline__ float sigm(float v) {
    return 1.0f / (1.0f + __expf(-v));
}

// A fragment (16x32 bf16, M x K): lanes 0-15 -> M=lane, K in {kb..kb+7, kb+16..kb+23}
//                                 lanes 16-31 -> M=lane-16, K in {kb+8..kb+15, kb+24..kb+31}
static __device__ __forceinline__ v16bf load_frag_a(const __bf16* buf, int rs, int m, int hi, int kbase) {
    union { v16bf v; uint4 q[2]; } u;
    const __bf16* p = buf + m * rs + kbase + hi * 8;
    u.q[0] = *(const uint4*)(p);
    u.q[1] = *(const uint4*)(p + 16);
    return u.v;
}

// B fragment (32x16 bf16, K x N), B = W^T with W row-major [N][K]:
// lane L -> column N = nbase + (L&15), K in {kbase+(L>>4)*8 .. +7, kbase+16+(L>>4)*8 .. +7}
static __device__ __forceinline__ v16bf load_frag_b(const __bf16* Wm, int ks, int n, int hi, int kbase) {
    union { v16bf v; uint4 q[2]; } u;
    const __bf16* p = Wm + n * ks + kbase + hi * 8;
    u.q[0] = *(const uint4*)(p);
    u.q[1] = *(const uint4*)(p + 16);
    return u.v;
}

// ---------------- prep: fold input projection into LSTM input weights ----------------
__global__ __launch_bounds__(256) void prep_weights(
        const float* __restrict__ W_in, const float* __restrict__ b_in,
        const float* __restrict__ W_ih, const float* __restrict__ b_ih,
        const float* __restrict__ W_hh, const float* __restrict__ b_hh,
        __bf16* __restrict__ wsWeff, __bf16* __restrict__ wsWhh,
        float* __restrict__ wsBeff) {
    int idx = blockIdx.x * 256 + threadIdx.x;
    if (idx < GATES * DPAD) {
        int n = idx / DPAD, d = idx - n * DPAD;
        float acc = 0.0f;
        if (d < DIN) {
            for (int e = 0; e < 256; ++e)
                acc += W_ih[n * 256 + e] * W_in[e * DIN + d];
        }
        wsWeff[idx] = (__bf16)acc;
    } else if (idx < GATES * DPAD + GATES * HID) {
        int j = idx - GATES * DPAD;
        wsWhh[j] = (__bf16)W_hh[j];
    } else if (idx < GATES * DPAD + GATES * HID + GATES) {
        int n = idx - (GATES * DPAD + GATES * HID);
        float acc = b_ih[n] + b_hh[n];
        for (int e = 0; e < 256; ++e)
            acc += W_ih[n * 256 + e] * b_in[e];
        wsBeff[n] = acc;
    }
}

// ---------------- Gx pre-GEMM: Gx[bt][s] tile (16 x 512) = x_tile @ W_eff^T ----------------
// Stored bf16 in WMMA C/D lane-swizzled order: [((bt*SEQ+s)*32 + ntile)*256 + lane*8 + r]
__global__ __launch_bounds__(256) void gx_gemm(
        const float* __restrict__ x,
        const __bf16* __restrict__ gWeff,
        __bf16* __restrict__ gx) {
    extern __shared__ char smem[];
    __bf16* Weff = (__bf16*)(smem + GL_WEFF);   // [512][96]
    __bf16* xt   = (__bf16*)(smem + GL_XT);     // [16][96]

    const int t    = threadIdx.x;
    const int lane = t & 31;
    const int wv   = t >> 5;
    const int l15  = lane & 15;
    const int hi   = lane >> 4;
    const int bt   = blockIdx.x >> 9;     // 0..15
    const int s    = blockIdx.x & (SEQ - 1);
    const int b0   = bt * 16;

    for (int i = t; i < GATES * DPAD; i += 256) Weff[i] = gWeff[i];
    // x tile (pad columns zeroed; 256 pad elems = one per thread, disjoint from data writes)
    xt[(t >> 4) * DPAD + DIN + (t & 15)] = (__bf16)0.0f;
    #pragma unroll
    for (int u = 0; u < 5; ++u) {
        int e = t + u * 256;
        int r = e / DIN, d = e - r * DIN;
        xt[r * DPAD + d] = (__bf16)x[(((long)(b0 + r)) * SEQ + s) * DIN + d];
    }
    __syncthreads();

    v16bf ax0 = load_frag_a(xt, DPAD, l15, hi, 0);
    v16bf ax1 = load_frag_a(xt, DPAD, l15, hi, 32);
    v16bf ax2 = load_frag_a(xt, DPAD, l15, hi, 64);

    #pragma unroll
    for (int i = 0; i < 4; ++i) {
        const int nt = wv * 4 + i;
        const int n  = nt * 16 + l15;
        v8f acc = {0.f, 0.f, 0.f, 0.f, 0.f, 0.f, 0.f, 0.f};
        acc = __builtin_amdgcn_wmma_f32_16x16x32_bf16(false, ax0, false, load_frag_b(Weff, DPAD, n, hi, 0),  (short)0, acc, false, false);
        acc = __builtin_amdgcn_wmma_f32_16x16x32_bf16(false, ax1, false, load_frag_b(Weff, DPAD, n, hi, 32), (short)0, acc, false, false);
        acc = __builtin_amdgcn_wmma_f32_16x16x32_bf16(false, ax2, false, load_frag_b(Weff, DPAD, n, hi, 64), (short)0, acc, false, false);
        union { uint4 q; __bf16 h[8]; } o;
        #pragma unroll
        for (int r = 0; r < 8; ++r) o.h[r] = (__bf16)acc[r];
        *(uint4*)(gx + (((size_t)(bt * SEQ + s)) * 32 + nt) * 256 + lane * 8) = o.q;
    }
}

// ---------------- recurrent kernel: 16 batch rows per WG, loop over S steps ----------------
template <bool USE_GX>
__global__ __launch_bounds__(256) void lstm_recurrent(
        const float* __restrict__ x,
        const __bf16* __restrict__ gGx,
        const __bf16* __restrict__ gWeff, const __bf16* __restrict__ gWhh,
        const float* __restrict__ gBeff,
        const float* __restrict__ gWtime,
        const float* __restrict__ gWbr, const float* __restrict__ gBbr,
        float* __restrict__ out) {
    extern __shared__ char smem[];
    __bf16* Whh   = (__bf16*)(smem + L_WHH);    // [512][128]
    float*  gates = (float*)(smem + L_GATES);   // [16][512]
    __bf16* hbuf  = (__bf16*)(smem + L_HBUF);   // [16][128]
    float*  bias  = (float*)(smem + L_BIAS);    // [512]
    float*  wbr   = (float*)(smem + L_WBR);     // [4][128]
    float*  misc  = (float*)(smem + L_MISC);    // [8]: wt0..2, bbr0..3
    __bf16* Weff  = (__bf16*)(smem + L_WEFF);   // fused only
    __bf16* xbuf  = (__bf16*)(smem + L_XBUF);   // fused only

    const int t    = threadIdx.x;
    const int lane = t & 31;
    const int wv   = t >> 5;       // wave id 0..7
    const int l15  = lane & 15;
    const int hi   = lane >> 4;
    const int row  = t >> 4;       // 0..15 (elementwise row)
    const int jb   = t & 15;       // 0..15 (elementwise column base)
    const int bt   = blockIdx.x;
    const int b0   = bt * 16;

    // -------- one-time LDS init --------
    for (int i = t; i < GATES * HID;  i += 256) Whh[i]  = gWhh[i];
    for (int i = t; i < GATES;        i += 256) bias[i] = gBeff[i];
    for (int i = t; i < 4 * HID;      i += 256) wbr[i]  = gWbr[i];
    for (int i = t; i < 16 * HID;     i += 256) hbuf[i] = (__bf16)0.0f;
    if (!USE_GX) {
        for (int i = t; i < GATES * DPAD;  i += 256) Weff[i] = gWeff[i];
        for (int i = t; i < 2 * 16 * DPAD; i += 256) xbuf[i] = (__bf16)0.0f;
    }
    if (t < 3) misc[t]     = gWtime[t];
    if (t < 4) misc[4 + t] = gBbr[t];
    __syncthreads();

    // stage step-0 inputs
    uint4 gxr[4];
    if (USE_GX) {
        #pragma unroll
        for (int i = 0; i < 4; ++i)
            gxr[i] = *(const uint4*)(gGx + (((size_t)bt * SEQ + 0) * 32 + (wv * 4 + i)) * 256 + lane * 8);
    } else {
        #pragma unroll
        for (int u = 0; u < 5; ++u) {
            int e = t + u * 256;
            int r = e / DIN, d = e - r * DIN;
            xbuf[r * DPAD + d] = (__bf16)x[(((long)(b0 + r)) * SEQ) * DIN + d];
        }
    }
    const float wt0 = misc[0], wt1 = misc[1], wt2 = misc[2];
    const float bb0 = misc[4], bb1 = misc[5], bb2 = misc[6], bb3 = misc[7];
    __syncthreads();

    // per-thread recurrent state (8 hidden elems each: j = jb + k*16)
    float creg[8] = {0, 0, 0, 0, 0, 0, 0, 0};
    float hst0[8] = {0}, hst1[8] = {0};
    float cst0[8] = {0}, cst1[8] = {0};

    for (int s = 0; s < SEQ; ++s) {
        const int cur = s & 1;
        const bool haveNext = (s + 1 < SEQ);

        // register double-buffer of the next step's input-gate contribution
        uint4 gxn[4];
        float xn[5];
        if (USE_GX) {
            if (haveNext) {
                #pragma unroll
                for (int i = 0; i < 4; ++i)
                    gxn[i] = *(const uint4*)(gGx + (((size_t)bt * SEQ + (s + 1)) * 32 + (wv * 4 + i)) * 256 + lane * 8);
            }
            if (s + 2 < SEQ)
                __builtin_prefetch(gGx + (((size_t)bt * SEQ + (s + 2)) * 32 + (wv * 4)) * 256 + lane * 8, 0, 1);
        } else {
            if (haveNext) {
                #pragma unroll
                for (int u = 0; u < 5; ++u) {
                    int e = t + u * 256;
                    int r = e / DIN, d = e - r * DIN;
                    xn[u] = x[(((long)(b0 + r)) * SEQ + (s + 1)) * DIN + d];
                }
            }
            if (s + 2 < SEQ)
                __builtin_prefetch(&x[(((long)(b0 + row)) * SEQ + (s + 2)) * DIN + jb * 5], 0, 1);
        }

        // A fragments shared across the 4 N-tiles this wave owns
        v16bf ah0 = load_frag_a(hbuf, HID, l15, hi, 0);
        v16bf ah1 = load_frag_a(hbuf, HID, l15, hi, 32);
        v16bf ah2 = load_frag_a(hbuf, HID, l15, hi, 64);
        v16bf ah3 = load_frag_a(hbuf, HID, l15, hi, 96);
        v16bf ax0, ax1, ax2;
        if (!USE_GX) {
            const __bf16* xc = xbuf + cur * (16 * DPAD);
            ax0 = load_frag_a(xc, DPAD, l15, hi, 0);
            ax1 = load_frag_a(xc, DPAD, l15, hi, 32);
            ax2 = load_frag_a(xc, DPAD, l15, hi, 64);
        }

        #pragma unroll
        for (int i = 0; i < 4; ++i) {
            const int nb = (wv * 4 + i) * 16;
            const int n  = nb + l15;
            v8f acc;
            if (USE_GX) {
                union { uint4 q; __bf16 h[8]; } u;
                u.q = gxr[i];
                #pragma unroll
                for (int r = 0; r < 8; ++r) acc[r] = (float)u.h[r];
            } else {
                acc = (v8f){0.f, 0.f, 0.f, 0.f, 0.f, 0.f, 0.f, 0.f};
                acc = __builtin_amdgcn_wmma_f32_16x16x32_bf16(false, ax0, false, load_frag_b(Weff, DPAD, n, hi, 0),  (short)0, acc, false, false);
                acc = __builtin_amdgcn_wmma_f32_16x16x32_bf16(false, ax1, false, load_frag_b(Weff, DPAD, n, hi, 32), (short)0, acc, false, false);
                acc = __builtin_amdgcn_wmma_f32_16x16x32_bf16(false, ax2, false, load_frag_b(Weff, DPAD, n, hi, 64), (short)0, acc, false, false);
            }
            acc = __builtin_amdgcn_wmma_f32_16x16x32_bf16(false, ah0, false, load_frag_b(Whh, HID, n, hi, 0),  (short)0, acc, false, false);
            acc = __builtin_amdgcn_wmma_f32_16x16x32_bf16(false, ah1, false, load_frag_b(Whh, HID, n, hi, 32), (short)0, acc, false, false);
            acc = __builtin_amdgcn_wmma_f32_16x16x32_bf16(false, ah2, false, load_frag_b(Whh, HID, n, hi, 64), (short)0, acc, false, false);
            acc = __builtin_amdgcn_wmma_f32_16x16x32_bf16(false, ah3, false, load_frag_b(Whh, HID, n, hi, 96), (short)0, acc, false, false);
            // C/D layout: VGPR r, lane L -> M = r + (L>>4)*8, N = nb + (L&15)
            #pragma unroll
            for (int r = 0; r < 8; ++r)
                gates[(r + hi * 8) * GATES + nb + l15] = acc[r];
        }
        __syncthreads();

        // -------- elementwise LSTM + T=3 stack pooling + fused 128->4 head --------
        const int slot = s % 3;
        float p0 = 0.f, p1 = 0.f, p2 = 0.f, p3 = 0.f;
        const float* gr = gates + row * GATES;
        #pragma unroll
        for (int k = 0; k < 8; ++k) {
            const int j = jb + k * 16;
            float ig = gr[j]       + bias[j];
            float fg = gr[j + 128] + bias[j + 128];
            float gg = gr[j + 256] + bias[j + 256];
            float og = gr[j + 384] + bias[j + 384];
            float cn = sigm(fg) * creg[k] + sigm(ig) * tanhf(gg);
            float hn = sigm(og) * tanhf(cn);
            float ho, co;
            if (slot == 0)      { hst0[k] = hn; cst0[k] = cn; ho = hn; co = cn; }
            else if (slot == 1) { hst1[k] = hn; cst1[k] = cn; ho = hn; co = cn; }
            else {
                ho = hst0[k] * wt0 + hst1[k] * wt1 + hn * wt2;
                co = cst0[k] * wt0 + cst1[k] * wt1 + cn * wt2;
                hst0[k] = 0.f; hst1[k] = 0.f; cst0[k] = 0.f; cst1[k] = 0.f;
            }
            creg[k] = co;
            hbuf[row * HID + j] = (__bf16)ho;           // next step's A matrix
            p0 += wbr[j] * ho;
            p1 += wbr[128 + j] * ho;
            p2 += wbr[256 + j] * ho;
            p3 += wbr[384 + j] * ho;
        }
        // butterfly-reduce the 4 logits across the 16 lanes owning this row
        #pragma unroll
        for (int off = 1; off < 16; off <<= 1) {
            p0 += __shfl_xor(p0, off);
            p1 += __shfl_xor(p1, off);
            p2 += __shfl_xor(p2, off);
            p3 += __shfl_xor(p3, off);
        }
        if (jb == 0) {
            float l0 = p0 + bb0, l1 = p1 + bb1, l2 = p2 + bb2, l3 = p3 + bb3;
            float mx  = fmaxf(fmaxf(l0, l1), fmaxf(l2, l3));
            float lse = logf(__expf(l0 - mx) + __expf(l1 - mx) + __expf(l2 - mx) + __expf(l3 - mx));
            float* o = out + (((long)(b0 + row)) * SEQ + s) * 4;
            o[0] = l0 - mx - lse; o[1] = l1 - mx - lse;
            o[2] = l2 - mx - lse; o[3] = l3 - mx - lse;
        }
        if (haveNext) {
            if (USE_GX) {
                #pragma unroll
                for (int i = 0; i < 4; ++i) gxr[i] = gxn[i];
            } else {
                __bf16* xnb = xbuf + (cur ^ 1) * (16 * DPAD);
                #pragma unroll
                for (int u = 0; u < 5; ++u) {
                    int e = t + u * 256;
                    int r = e / DIN, d = e - r * DIN;
                    xnb[r * DPAD + d] = (__bf16)xn[u];
                }
            }
        }
        __syncthreads();
    }
}

extern "C" void kernel_launch(void* const* d_in, const int* in_sizes, int n_in,
                              void* d_out, int out_size, void* d_ws, size_t ws_size,
                              hipStream_t stream) {
    const float* x     = (const float*)d_in[0];
    const float* W_in  = (const float*)d_in[1];
    const float* b_in  = (const float*)d_in[2];
    const float* W_ih  = (const float*)d_in[3];
    const float* b_ih  = (const float*)d_in[4];
    const float* W_hh  = (const float*)d_in[5];
    const float* b_hh  = (const float*)d_in[6];
    const float* wtime = (const float*)d_in[7];
    const float* W_br  = (const float*)d_in[8];
    const float* b_br  = (const float*)d_in[9];

    __bf16* wsWeff = (__bf16*)((char*)d_ws + WS_WEFF);
    __bf16* wsWhh  = (__bf16*)((char*)d_ws + WS_WHH);
    float*  wsBeff = (float*)((char*)d_ws + WS_BEFF);
    __bf16* wsGx   = (__bf16*)((char*)d_ws + WS_GX);

    const bool useGx = (ws_size >= (size_t)WS_GX + GX_BYTES);

    int prep_items = GATES * DPAD + GATES * HID + GATES;
    prep_weights<<<(prep_items + 255) / 256, 256, 0, stream>>>(
        W_in, b_in, W_ih, b_ih, W_hh, b_hh, wsWeff, wsWhh, wsBeff);

    if (useGx) {
        (void)hipFuncSetAttribute((const void*)gx_gemm,
                                  hipFuncAttributeMaxDynamicSharedMemorySize, SMEM_GEMM);
        gx_gemm<<<dim3(16 * SEQ), dim3(256), SMEM_GEMM, stream>>>(x, wsWeff, wsGx);

        (void)hipFuncSetAttribute((const void*)lstm_recurrent<true>,
                                  hipFuncAttributeMaxDynamicSharedMemorySize, SMEM_GX);
        lstm_recurrent<true><<<dim3(BSZ / 16), dim3(256), SMEM_GX, stream>>>(
            x, wsGx, wsWeff, wsWhh, wsBeff, wtime, W_br, b_br, (float*)d_out);
    } else {
        (void)hipFuncSetAttribute((const void*)lstm_recurrent<false>,
                                  hipFuncAttributeMaxDynamicSharedMemorySize, SMEM_FULL);
        lstm_recurrent<false><<<dim3(BSZ / 16), dim3(256), SMEM_FULL, stream>>>(
            x, nullptr, wsWeff, wsWhh, wsBeff, wtime, W_br, b_br, (float*)d_out);
    }
}
